// MultiHeadAttention_4655744549069
// MI455X (gfx1250) — matrix operand order repaired and checked by reading, not compiled, not run on hardware
//
#include <hip/hip_runtime.h>

// ---------------------------------------------------------------------------
// MI455X (gfx1250) multi-head attention forward.
// fp32 reference -> bf16 WMMA (V_WMMA_F32_16X16X32_BF16) with f32 accumulate.
// Flash-style attention (online softmax). K tiles DMA'd into LDS by the
// Tensor Data Mover, double-buffered so the DMA of tile i+1 overlaps the
// WMMAs of tile i. Softmax row-sums are computed on the MATRIX pipe
// (P x ones via WMMA) instead of cross-lane butterflies, and exps are base-2
// (log2e folded into the Q pre-scale along with 1/sqrt(d)).
// ---------------------------------------------------------------------------

#define S_LEN 4096
#define EMB   512
#define NH    8
#define HD    64
#define QKV_LD 1536   // qkv row stride: [q(512) | k(512) | v(512)]

// padded LDS row strides (elements): keep 16B alignment, rotate banks
#define GPAD 40       // GEMM tiles: 32 K + 8 pad
#define APAD 72       // attention tiles: 64 + 8 pad
#define KTILE (64 * APAD)   // one attention tile in elements (9216 B)

// 1/sqrt(64) * log2(e): scores come out ready for exp2
#define QSCALE (0.125f * 1.44269504088896f)

typedef __bf16 bf16_t;
typedef __attribute__((ext_vector_type(16))) __bf16        v16bf;
typedef __attribute__((ext_vector_type(8)))  float         v8f;
typedef __attribute__((ext_vector_type(4)))  unsigned int  u32x4;
typedef __attribute__((ext_vector_type(4)))  int           i32x4;
typedef __attribute__((ext_vector_type(8)))  int           i32x8;

union Frag16 { v16bf v; u32x4 q[2]; };

__device__ inline bf16_t cvt_bf16(float  x) { return (bf16_t)x; }
__device__ inline bf16_t cvt_bf16(bf16_t x) { return x; }

__device__ inline v8f v8f_zero() {
  v8f z = {0.f, 0.f, 0.f, 0.f, 0.f, 0.f, 0.f, 0.f};
  return z;
}

// store 16 bf16 (two ds_store_b128)
__device__ inline void st16(bf16_t* dst, const bf16_t* vals) {
  *(u32x4*)(dst)     = *(const u32x4*)(vals);
  *(u32x4*)(dst + 8) = *(const u32x4*)(vals + 8);
}

// ---------------------------------------------------------------------------
// TDM: DMA one 64-row x 64-element bf16 tile (row stride QKV_LD elements in
// global) into LDS at lds_off, padding each 128B row with 16B so LDS rows
// land APAD(=72) elements apart. D# layout per CDNA5 ISA ch.8.
// ---------------------------------------------------------------------------
__device__ inline void tdm_load_k_tile(unsigned lds_off, unsigned long long ga)
{
  u32x4 g0;
  g0.x = 1u;                                          // count=1 valid user D#
  g0.y = lds_off;                                     // lds_addr (bytes)
  g0.z = (unsigned)(ga & 0xFFFFFFFFull);              // global_addr[31:0]
  g0.w = (unsigned)((ga >> 32) & 0x1FFFFFFull)        // global_addr[56:32]
       | (2u << 30);                                  // type = 2 ("image")

  i32x8 g1;
  g1[0] = (1 << 16)        // data_size = 1 -> 2 bytes
        | (1 << 20)        // pad_enable
        | (4 << 22)        // pad_interval: 2^(4+1)=32 DWORDs = 128B (one row)
        | (3 << 25);       // pad_amount: 4 DWORDs = 16B = 8 elements
  g1[1] = (64 << 16);      // tensor_dim0[15:0]=64 (bits 63:48)
  g1[2] = (64 << 16);      // tensor_dim1[15:0]=64 (bits 95:80)
  g1[3] = (64 << 16);      // tile_dim0=64        (bits 127:112)
  g1[4] = 64;              // tile_dim1=64, tile_dim2=0
  g1[5] = QKV_LD;          // tensor_dim0_stride[31:0]
  g1[6] = 0;
  g1[7] = 0;

  i32x4 z4 = {0, 0, 0, 0};
#if __clang_major__ >= 23
  i32x8 z8 = {0, 0, 0, 0, 0, 0, 0, 0};
  __builtin_amdgcn_tensor_load_to_lds(g0, g1, z4, z4, z8, 0);
#else
  __builtin_amdgcn_tensor_load_to_lds(g0, g1, z4, z4, 0);
#endif
}

// ---------------------------------------------------------------------------
// Generic GEMM:  C[M,N] = (A[M,K] * B[K,N] + bias[N]) * (n < q_cols ? q_scale : 1)
// Block tile 128x128x32, 256 threads = 8 waves, wave tile 32x64 (2x4 WMMA).
// Register-prefetch pipeline: next A/B tiles are loaded from global *before*
// the WMMA block so memory latency hides under matrix ops.
// ---------------------------------------------------------------------------
template <typename AT>
__device__ inline void gemm_load_a(bf16_t* vals, const AT* A, int bm, int row,
                                   int col, int k0, int K) {
  const AT* src = A + (size_t)(bm + row) * K + k0 + col;
#pragma unroll
  for (int e = 0; e < 16; ++e) vals[e] = cvt_bf16(src[e]);
}
__device__ inline void gemm_load_b(bf16_t* vals, const float* B, int bn, int n,
                                   int kh, int k0, int N) {
#pragma unroll
  for (int e = 0; e < 16; ++e)
    vals[e] = cvt_bf16(B[(size_t)(k0 + kh * 16 + e) * N + bn + n]);
}

template <typename AT, typename CT>
__global__ __launch_bounds__(256)
void gemm_bias_wmma(const AT* __restrict__ A, const float* __restrict__ B,
                    const float* __restrict__ bias, CT* __restrict__ C,
                    int M, int N, int K, float q_scale, int q_cols)
{
  __shared__ bf16_t sA[128 * GPAD];   // [m][k]
  __shared__ bf16_t sB[128 * GPAD];   // [n][k]

  const int tid  = threadIdx.x;
  const int lane = tid & 31;
  const int w    = tid >> 5;
  const int wrow = w & 3;             // 4 wave-rows of 32 M each
  const int wcol = w >> 2;            // 2 wave-cols of 64 N each
  const int bm   = blockIdx.y * 128;
  const int bn   = blockIdx.x * 128;

  const int arow = (tid * 16) >> 5;   // A staging coords (contiguous copy)
  const int acol = (tid * 16) & 31;
  const int nb   = tid & 127;         // B staging: column gather (coalesced)
  const int kh   = tid >> 7;

  v8f acc[2][4];
#pragma unroll
  for (int i = 0; i < 2; ++i)
#pragma unroll
    for (int j = 0; j < 4; ++j) acc[i][j] = v8f_zero();

  // prologue: stage tile k0=0
  bf16_t av[16], bv[16];
  gemm_load_a(av, A, bm, arow, acol, 0, K);
  gemm_load_b(bv, B, bn, nb, kh, 0, N);
  st16(sA + arow * GPAD + acol, av);
  st16(sB + nb * GPAD + kh * 16, bv);

  for (int k0 = 0; k0 < K; k0 += 32) {
    __syncthreads();
    const bool pre = (k0 + 32 < K);
    if (pre) {                         // prefetch next tile into registers
      gemm_load_a(av, A, bm, arow, acol, k0 + 32, K);
      gemm_load_b(bv, B, bn, nb, kh, k0 + 32, N);
    }

    // ---- WMMA fragments in ISA operand layout (ch.7.12.2) ----
    Frag16 af[2], bfg[4];
#pragma unroll
    for (int i = 0; i < 2; ++i) {
      int row = wrow * 32 + i * 16 + (lane & 15);
      int kb  = (lane >> 4) * 8;      // A-frag: K chunks at kb and kb+16
      const bf16_t* p = sA + row * GPAD + kb;
      af[i].q[0] = *(const u32x4*)(p);
      af[i].q[1] = *(const u32x4*)(p + 16);
    }
#pragma unroll
    for (int j = 0; j < 4; ++j) {
      int col = wcol * 64 + j * 16 + (lane & 15);
      int kb  = (lane >> 4) * 8;      // B-frag: K chunks at kb and kb+16
      const bf16_t* p = sB + col * GPAD + kb;
      bfg[j].q[0] = *(const u32x4*)(p);
      bfg[j].q[1] = *(const u32x4*)(p + 16);
    }
#pragma unroll
    for (int i = 0; i < 2; ++i)
#pragma unroll
      for (int j = 0; j < 4; ++j)
        acc[i][j] = __builtin_amdgcn_wmma_f32_16x16x32_bf16(
            false, af[i].v, false, bfg[j].v, (short)0, acc[i][j], false, false);

    __syncthreads();
    if (pre) {
      st16(sA + arow * GPAD + acol, av);
      st16(sB + nb * GPAD + kh * 16, bv);
    }
  }

  // ---- epilogue: D-matrix layout -> global, optional Q pre-scale ----
#pragma unroll
  for (int i = 0; i < 2; ++i) {
    int mrow = bm + wrow * 32 + i * 16 + (lane >> 4) * 8;
#pragma unroll
    for (int j = 0; j < 4; ++j) {
      int n = bn + wcol * 64 + j * 16 + (lane & 15);
      float bvv = bias[n];
      float s   = (n < q_cols) ? q_scale : 1.0f;
#pragma unroll
      for (int r = 0; r < 8; ++r) {
        float v = (acc[i][j][r] + bvv) * s;
        C[(size_t)(mrow + r) * N + n] = (CT)v;
      }
    }
  }
}

// ---------------------------------------------------------------------------
// Flash attention: grid (S/128, H, B), 256 threads (8 waves), dynamic LDS:
//   [0)       sK[2]  2 x 64 x APAD bf16  (TDM double buffer, offsets 0/9216)
//   [18432B)  sV[2]  2 x 64 x APAD bf16  (V^T double buffer)
//   [36864B)  sP     8 waves x 16 x APAD bf16 (probs)
// Pipeline: at iteration i, wave0 issues the TDM for tile i+1 and all threads
// issue the V gather loads for i+1 BEFORE computing tile i.
// Scores arrive pre-scaled by 1/sqrt(d)*log2(e); softmax uses exp2 and the
// row-sum is computed as (P x ones) on the matrix pipe.
// ---------------------------------------------------------------------------
__device__ inline void gather_v(const bf16_t* qkv, int b, int h, int kv0,
                                int d, int kc, bf16_t* vals) {
  const bf16_t* src =
      qkv + (size_t)(b * S_LEN + kv0 + kc * 16) * QKV_LD + 2 * EMB + h * HD + d;
#pragma unroll
  for (int e = 0; e < 16; ++e) vals[e] = src[(size_t)e * QKV_LD];
}

__global__ __launch_bounds__(256)
void flash_attn_wmma(const bf16_t* __restrict__ qkv, bf16_t* __restrict__ out)
{
  extern __shared__ bf16_t smem[];
  bf16_t* sK = smem;                       // two buffers of [key][d]
  bf16_t* sV = smem + 2 * KTILE;           // two buffers of [d][key]
  bf16_t* sP = sV   + 2 * KTILE;           // per-wave [m][key]

  const int tid  = threadIdx.x;
  const int lane = tid & 31;
  const int w    = tid >> 5;
  const int b    = blockIdx.z;
  const int h    = blockIdx.y;
  const int q0   = blockIdx.x * 128 + w * 16;

  const int vd   = tid & 63;               // V staging coords
  const int vkc  = tid >> 6;

  // Q fragments: resident in VGPRs for the whole KV loop (16 x 64 bf16)
  Frag16 qf[2];
  {
    const bf16_t* p =
        qkv + (size_t)(b * S_LEN + q0 + (lane & 15)) * QKV_LD + h * HD;
#pragma unroll
    for (int ks = 0; ks < 2; ++ks) {
      int kb = ks * 32 + (lane >> 4) * 8;
      qf[ks].q[0] = *(const u32x4*)(p + kb);
      qf[ks].q[1] = *(const u32x4*)(p + kb + 16);
    }
  }

  // constant all-ones B fragment (64x16 of 1.0 across the two K-steps)
  Frag16 ones;
#pragma unroll
  for (int e = 0; e < 16; ++e) ones.v[e] = (bf16_t)1.0f;

  float rowmax[8], rowsum[8];
  v8f accO[4];
#pragma unroll
  for (int r = 0; r < 8; ++r) { rowmax[r] = -3.0e38f; rowsum[r] = 0.f; }
#pragma unroll
  for (int dt = 0; dt < 4; ++dt) accO[dt] = v8f_zero();

  bf16_t* myP = sP + w * 16 * APAD;
  const size_t kbase = (size_t)(b * S_LEN) * QKV_LD + EMB + h * HD;

  // ---- prologue: stage tile 0 into buffer 0 ----
  if (w == 0) {
    tdm_load_k_tile(0u, (unsigned long long)(uintptr_t)(qkv + kbase));
  }
  {
    bf16_t vv[16];
    gather_v(qkv, b, h, 0, vd, vkc, vv);
    st16(sV + vd * APAD + vkc * 16, vv);
  }
  if (w == 0) __builtin_amdgcn_s_wait_tensorcnt(0);
  __syncthreads();

  const int NKV = S_LEN / 64;
  for (int it = 0; it < NKV; ++it) {
    const int buf = it & 1;
    bf16_t* kb_ = sK + buf * KTILE;
    bf16_t* vb_ = sV + buf * KTILE;
    const bool pre = (it + 1 < NKV);

    // ---- kick off next tile's data movement before computing this one ----
    if (w == 0 && pre) {
      unsigned long long ga = (unsigned long long)(uintptr_t)(
          qkv + kbase + (size_t)(it + 1) * 64 * QKV_LD);
      tdm_load_k_tile((unsigned)((buf ^ 1) * KTILE * 2), ga);  // byte offset
    }
    bf16_t vv[16];
    if (pre) gather_v(qkv, b, h, (it + 1) * 64, vd, vkc, vv);

    // ---- S = Q K^T : 4 key-subtiles x 2 K-steps of 32 ----
    v8f sacc[4];
#pragma unroll
    for (int j = 0; j < 4; ++j) sacc[j] = v8f_zero();
#pragma unroll
    for (int j = 0; j < 4; ++j) {
#pragma unroll
      for (int ks = 0; ks < 2; ++ks) {
        Frag16 kf;
        int col = j * 16 + (lane & 15);
        int kbo = ks * 32 + (lane >> 4) * 8;
        const bf16_t* p = kb_ + col * APAD + kbo;
        kf.q[0] = *(const u32x4*)(p);
        kf.q[1] = *(const u32x4*)(p + 16);
        sacc[j] = __builtin_amdgcn_wmma_f32_16x16x32_bf16(
            false, qf[ks].v, false, kf.v, (short)0, sacc[j], false, false);
      }
    }

    // ---- park prefetched V in the alternate buffer ----
    if (pre) st16(sV + (buf ^ 1) * KTILE + vd * APAD + vkc * 16, vv);

    // ---- online softmax: only the MAX needs cross-lane traffic; the sum is
    //      computed post-barrier on the matrix pipe. Base-2 exponentials. ----
    int mrow = (lane >> 4) * 8;
#pragma unroll
    for (int r = 0; r < 8; ++r) {
      float mloc = -3.0e38f;
#pragma unroll
      for (int j = 0; j < 4; ++j) mloc = fmaxf(mloc, sacc[j][r]);
#pragma unroll
      for (int o = 8; o > 0; o >>= 1)
        mloc = fmaxf(mloc, __shfl_xor(mloc, o, 16));
      float mnew  = fmaxf(rowmax[r], mloc);
      float alpha = __builtin_exp2f(rowmax[r] - mnew);
      rowmax[r] = mnew;
#pragma unroll
      for (int j = 0; j < 4; ++j) {
        float pv = __builtin_exp2f(sacc[j][r] - mnew);
        myP[(mrow + r) * APAD + j * 16 + (lane & 15)] = (bf16_t)pv;
      }
      rowsum[r] *= alpha;                 // deferred: += tile-sum after barrier
#pragma unroll
      for (int dt = 0; dt < 4; ++dt) accO[dt][r] *= alpha;
    }
    __syncthreads();   // P visible across lanes before A-frag reload

    // ---- P fragments (A-layout) for both sum and PV ----
    Frag16 pf[2];
#pragma unroll
    for (int ks = 0; ks < 2; ++ks) {
      int m   = lane & 15;
      int kbo = ks * 32 + (lane >> 4) * 8;
      const bf16_t* pp = myP + m * APAD + kbo;
      pf[ks].q[0] = *(const u32x4*)(pp);
      pf[ks].q[1] = *(const u32x4*)(pp + 16);
    }

    // row-sum = P x ones : lands broadcast across columns in D-layout, i.e.
    // directly in the lane/VGPR slot where rowsum[r] lives. No shuffles.
    v8f ssum = v8f_zero();
    ssum = __builtin_amdgcn_wmma_f32_16x16x32_bf16(
        false, pf[0].v, false, ones.v, (short)0, ssum, false, false);
    ssum = __builtin_amdgcn_wmma_f32_16x16x32_bf16(
        false, pf[1].v, false, ones.v, (short)0, ssum, false, false);
#pragma unroll
    for (int r = 0; r < 8; ++r) rowsum[r] += ssum[r];

    // ---- O += P V : 4 d-subtiles x 2 K-steps of 32 keys ----
#pragma unroll
    for (int ks = 0; ks < 2; ++ks) {
#pragma unroll
      for (int dt = 0; dt < 4; ++dt) {
        Frag16 vf;
        int col = dt * 16 + (lane & 15);
        int kb2 = ks * 32 + (lane >> 4) * 8;
        const bf16_t* vp2 = vb_ + col * APAD + kb2;
        vf.q[0] = *(const u32x4*)(vp2);
        vf.q[1] = *(const u32x4*)(vp2 + 16);
        accO[dt] = __builtin_amdgcn_wmma_f32_16x16x32_bf16(
            false, pf[ks].v, false, vf.v, (short)0, accO[dt], false, false);
      }
    }

    // next tile's DMA must be complete before anyone reads it after the
    // barrier; TDM is in-order per wave so tensorcnt==0 is sufficient.
    if (w == 0 && pre) __builtin_amdgcn_s_wait_tensorcnt(0);
    __syncthreads();
  }

  // ---- normalize, emit bf16 attn output [B*S, 512] ----
#pragma unroll
  for (int r = 0; r < 8; ++r) {
    float inv = 1.0f / rowsum[r];
    int m = q0 + (lane >> 4) * 8 + r;
    size_t rowoff = (size_t)(b * S_LEN + m) * EMB + h * HD;
#pragma unroll
    for (int dt = 0; dt < 4; ++dt) {
      int d = dt * 16 + (lane & 15);
      out[rowoff + d] = (bf16_t)(accO[dt][r] * inv);
    }
  }
}

// ---------------------------------------------------------------------------
extern "C" void kernel_launch(void* const* d_in, const int* in_sizes, int n_in,
                              void* d_out, int out_size, void* d_ws, size_t ws_size,
                              hipStream_t stream)
{
  const float* x     = (const float*)d_in[0];   // [2,4096,512]
  const float* w_qkv = (const float*)d_in[1];   // [512,1536]
  const float* b_qkv = (const float*)d_in[2];   // [1536]
  const float* w_out = (const float*)d_in[3];   // [512,512]
  const float* b_out = (const float*)d_in[4];   // [512]
  float* out = (float*)d_out;                   // [2,4096,512] f32

  const int B = 2;
  const int M = B * S_LEN;                      // 8192

  // workspace: qkv bf16 [8192,1536] (24 MB) + attn bf16 [8192,512] (8 MB)
  bf16_t* qkv  = (bf16_t*)d_ws;
  bf16_t* attn = (bf16_t*)((char*)d_ws + (size_t)M * QKV_LD * sizeof(bf16_t));

  // 1) QKV projection (fp32 in -> bf16 out); Q pre-scaled by 1/sqrt(d)*log2e
  gemm_bias_wmma<float, bf16_t>
      <<<dim3((3 * EMB) / 128, M / 128), 256, 0, stream>>>(
          x, w_qkv, b_qkv, qkv, M, 3 * EMB, EMB, QSCALE, EMB);

  // 2) flash attention per (b, h, 128-row Q tile)
  size_t smem_bytes = (size_t)(4 * KTILE + 8 * 16 * APAD) * sizeof(bf16_t);
  flash_attn_wmma<<<dim3(S_LEN / 128, NH, B), 256, smem_bytes, stream>>>(qkv, attn);

  // 3) output projection (bf16 in -> fp32 out)
  gemm_bias_wmma<bf16_t, float>
      <<<dim3(EMB / 128, M / 128), 256, 0, stream>>>(
          attn, w_out, b_out, out, M, EMB, EMB, 1.0f, 0);
}
